// MoRALayer_43508018709044
// MI455X (gfx1250) — matrix operand order repaired and checked by reading, not compiled, not run on hardware
//
#include <hip/hip_runtime.h>
#include <hip/hip_bf16.h>
#include <stdint.h>
#include <stddef.h>

// ---------------- problem constants ----------------
constexpr int Bsz = 8192, Dm = 1024, Hm = 1024, Om = 1024;
constexpr int Em = 8, Lm = 2, Km = 2, Tm = 4;

// ---------------- vector types ----------------
typedef __bf16  v16bf __attribute__((ext_vector_type(16)));
typedef __bf16  v8bf  __attribute__((ext_vector_type(8)));
typedef __bf16  v4bf  __attribute__((ext_vector_type(4)));
typedef float   v8f   __attribute__((ext_vector_type(8)));

__device__ __forceinline__ float sigmoidf_(float x) {
  return 1.0f / (1.0f + __expf(-x));
}

// CDNA5 async copy: global memory -> LDS, 16B per lane, tracked by ASYNCcnt.
__device__ __forceinline__ void async_b128_to_lds(uint32_t lds_addr, uint64_t gaddr) {
  asm volatile("global_load_async_to_lds_b128 %0, %1, off"
               :: "v"(lds_addr), "v"(gaddr) : "memory");
}
__device__ __forceinline__ void wait_asynccnt0() {
  asm volatile("s_wait_asynccnt 0x0" ::: "memory");
}

__device__ __forceinline__ v16bf frag_pair(const __bf16* p0, const __bf16* p1) {
  union { v16bf v; struct { v8bf lo; v8bf hi; } s; } u;
  u.s.lo = *(const v8bf*)p0;
  u.s.hi = *(const v8bf*)p1;
  return u.v;
}

// ---------------- fp32 -> bf16 convert ----------------
__global__ __launch_bounds__(256) void k_cvt_bf16(const float* __restrict__ s,
                                                  __bf16* __restrict__ d,
                                                  long long n4) {
  long long i = (long long)blockIdx.x * blockDim.x + threadIdx.x;
  long long stride = (long long)gridDim.x * blockDim.x;
  const float4* s4 = (const float4*)s;
  for (; i < n4; i += stride) {
    float4 v = s4[i];
    v4bf o;
    o[0] = (__bf16)v.x; o[1] = (__bf16)v.y; o[2] = (__bf16)v.z; o[3] = (__bf16)v.w;
    *(v4bf*)&d[i * 4] = o;
  }
}

// ---------------- WMMA GEMM: C[M,N] = act(A[M,K] * B + bias) ----------------
// Block tile 128x128x32, 256 threads = 8 waves in a 4(M) x 2(N) grid.
// Each wave owns a 32x64 strip -> 8x v_wmma_f32_16x16x32_bf16 per K-tile from
// 2 A-fragments + 4 B-fragments (best WMMA : ds_load ratio at this LDS size).
struct GemmArgs {
  const __bf16* A;        // bf16 activations
  const __bf16* Bm;       // bf16 weights
  const float*  bias;     // length N (per expert) or null
  void*         C;        // f32 or bf16
  const int*    map;      // row gather map (or null)
  const int*    counts;   // per-expert row count (or null)
  long long bExpStride;   // elements between experts in Bm
  long long biasExpStride;
  int mapStride;          // rows between experts in map
  int lda, ldb, ldc;
  int aShift;             // A row = map[g] >> aShift
  int bT;                 // 1: B global layout is [N,K] row-major, else [K,N]
  int cBf16, relu;
  int M, N, Kd;
};

#define TBM 128
#define TBN 128
#define TBK 32
#define LDS_STR 40   // 80 bytes/row: keeps 16B alignment, staggers banks

__global__ __launch_bounds__(256) void k_gemm_wmma(GemmArgs ga) {
  __shared__ alignas(16) __bf16 As[TBM][LDS_STR];
  __shared__ alignas(16) __bf16 Bs[TBN][LDS_STR];   // [n][k]

  const int tid    = threadIdx.x;
  const int lane   = tid & 31;
  const int wave   = tid >> 5;
  const int laneHi = lane >> 4;
  const int l15    = lane & 15;
  const int z      = blockIdx.z;

  int Mrows = ga.M;
  const __bf16* Bm  = ga.Bm;
  const float* bias = ga.bias;
  const int*   map  = ga.map;
  if (ga.counts) {
    Mrows = ga.counts[z];
    Bm   += (size_t)z * ga.bExpStride;
    if (bias) bias += (size_t)z * ga.biasExpStride;
    map  += (size_t)z * ga.mapStride;
  }
  const int rowBase = blockIdx.y * TBM;
  if (rowBase >= Mrows) return;
  const int colBase = blockIdx.x * TBN;

  const int m0 = (wave >> 1) * 32;   // 4 waves along M
  const int n0 = (wave & 1) * 64;    // 2 waves along N

  v8f acc[2][4] = {};

  // ---- A staging: thread t moves 32B of row (t>>1), k-cols [(t&1)*16 .. +15]
  const int arow = tid >> 1;
  const int acol = (tid & 1) * 16;
  int gRowA = rowBase + arow;
  int aSrcRow;
  if (map) {
    int g = (gRowA < Mrows) ? map[gRowA] : map[rowBase];
    aSrcRow = g >> ga.aShift;
  } else {
    aSrcRow = (gRowA < Mrows) ? gRowA : rowBase;
  }
  const __bf16* Aptr = ga.A + (size_t)aSrcRow * ga.lda + acol;
  const uint32_t ldsA0 = (uint32_t)(uintptr_t)&As[arow][acol];
  const uint32_t ldsA1 = (uint32_t)(uintptr_t)&As[arow][acol + 8];

  // ---- B staging indices ----
  // bT=1: thread t moves 32B of weight row (colBase + (t>>1)), k [(t&1)*16..]
  const int bn_t  = tid >> 1;
  const int bk_t  = (tid & 1) * 16;
  const uint32_t ldsB0 = (uint32_t)(uintptr_t)&Bs[bn_t][bk_t];
  const uint32_t ldsB1 = (uint32_t)(uintptr_t)&Bs[bn_t][bk_t + 8];
  // bT=0: thread t transposes 16 n at k = (t>>3), n0 = ((2t)&15)*8
  const int bk_n  = tid >> 3;
  const int bn_n  = ((2 * tid) & 15) * 8;

  for (int k0 = 0; k0 < ga.Kd; k0 += TBK) {
    // ---- stage A (128x32) via CDNA5 async copies ----
    async_b128_to_lds(ldsA0, (uint64_t)(uintptr_t)(Aptr + k0));
    async_b128_to_lds(ldsA1, (uint64_t)(uintptr_t)(Aptr + k0 + 8));
    // ---- stage B (32x128) into Bs[n][k] ----
    if (ga.bT) {  // B global [N,K]: K-contiguous, async straight to LDS
      const __bf16* bp = Bm + (size_t)(colBase + bn_t) * ga.ldb + k0 + bk_t;
      async_b128_to_lds(ldsB0, (uint64_t)(uintptr_t)bp);
      async_b128_to_lds(ldsB1, (uint64_t)(uintptr_t)(bp + 8));
    } else {      // B global [K,N]: transpose while staging
      const __bf16* bp = Bm + (size_t)(k0 + bk_n) * ga.ldb + colBase + bn_n;
      v8bf t0 = *(const v8bf*)bp;
      v8bf t1 = *(const v8bf*)(bp + 8);
#pragma unroll
      for (int j = 0; j < 8; ++j) {
        Bs[bn_n + j][bk_n]     = t0[j];
        Bs[bn_n + 8 + j][bk_n] = t1[j];
      }
    }
    if (k0 + TBK < ga.Kd) {
      __builtin_prefetch(Aptr + k0 + TBK, 0, 1);   // global_prefetch_b8
    }
    wait_asynccnt0();
    __syncthreads();

    // ---- fragments per CDNA5 wave32 layouts ----
    v16bf aF[2], bF[4];
#pragma unroll
    for (int mi = 0; mi < 2; ++mi) {
      const __bf16* ar = &As[m0 + mi * 16 + l15][0];
      aF[mi] = frag_pair(ar + laneHi * 8, ar + 16 + laneHi * 8);
    }
#pragma unroll
    for (int ni = 0; ni < 4; ++ni) {
      const __bf16* br = &Bs[n0 + ni * 16 + l15][0];
      bF[ni] = frag_pair(br + laneHi * 8, br + 16 + laneHi * 8);
    }
#pragma unroll
    for (int mi = 0; mi < 2; ++mi)
#pragma unroll
      for (int ni = 0; ni < 4; ++ni)
        acc[mi][ni] = __builtin_amdgcn_wmma_f32_16x16x32_bf16(
            false, aF[mi], false, bF[ni], (short)0, acc[mi][ni], false, false);
    __syncthreads();
  }

  // ---- epilogue: C layout M = laneHi*8 + r, N = lane&15 ----
#pragma unroll
  for (int mi = 0; mi < 2; ++mi) {
#pragma unroll
    for (int r = 0; r < 8; ++r) {
      int mTile = m0 + mi * 16 + laneHi * 8 + r;
      int gRow  = rowBase + mTile;
      if (gRow >= Mrows) continue;
      int cRow = map ? map[gRow] : gRow;
#pragma unroll
      for (int ni = 0; ni < 4; ++ni) {
        int nG = colBase + n0 + ni * 16 + l15;
        float v = acc[mi][ni][r];
        if (bias) v += bias[nG];
        if (ga.relu) v = fmaxf(v, 0.f);
        if (ga.cBf16) {
          ((__bf16*)ga.C)[(size_t)cRow * ga.ldc + nG] = (__bf16)v;
        } else {
          ((float*)ga.C)[(size_t)cRow * ga.ldc + nG] = v;
        }
      }
    }
  }
}

// ---------------- LSTM cell elementwise ----------------
__global__ __launch_bounds__(256) void k_lstm_cell(
    const float* __restrict__ zX, const float* __restrict__ zH,
    const float* __restrict__ bih, const float* __restrict__ bhh,
    float* __restrict__ c, float* __restrict__ h,
    __bf16* __restrict__ hbf, float* __restrict__ hsum) {
  int i = blockIdx.x * blockDim.x + threadIdx.x;
  if (i >= Bsz * Hm) return;
  int b = i >> 10, j = i & (Hm - 1);
  size_t base = (size_t)b * (4 * Hm);
  float zi = zX[base + j]          + zH[base + j]          + bih[j]          + bhh[j];
  float zf = zX[base + Hm + j]     + zH[base + Hm + j]     + bih[Hm + j]     + bhh[Hm + j];
  float zg = zX[base + 2 * Hm + j] + zH[base + 2 * Hm + j] + bih[2 * Hm + j] + bhh[2 * Hm + j];
  float zo = zX[base + 3 * Hm + j] + zH[base + 3 * Hm + j] + bih[3 * Hm + j] + bhh[3 * Hm + j];
  float cn = sigmoidf_(zf) * c[i] + sigmoidf_(zi) * tanhf(zg);
  float hn = sigmoidf_(zo) * tanhf(cn);
  c[i] = cn; h[i] = hn; hbf[i] = (__bf16)hn; hsum[i] += hn;
}

// ---------------- halt GEMV + ACT state ----------------
__global__ __launch_bounds__(256) void k_halt(
    const float* __restrict__ h, const float* __restrict__ halt_w,
    const float* __restrict__ halt_b, float* __restrict__ halt,
    float* __restrict__ rem) {
  int b = blockIdx.x * 8 + (threadIdx.x >> 5);
  int lane = threadIdx.x & 31;
  if (b >= Bsz) return;
  float s = 0.f;
  for (int j = lane; j < Hm; j += 32) s += h[(size_t)b * Hm + j] * halt_w[j];
#pragma unroll
  for (int off = 16; off; off >>= 1) s += __shfl_xor(s, off, 32);
  if (lane == 0) {
    float y = sigmoidf_(s + halt_b[0]);
    float ha = halt[b];
    ha = ha + y * (1.f - ha);
    halt[b] = ha;
    rem[b] += (1.f - ha);
  }
}

// ---------------- out = h_sum * rem / T ----------------
__global__ __launch_bounds__(256) void k_act_final(
    const float* __restrict__ hsum, const float* __restrict__ rem,
    float* __restrict__ outp, __bf16* __restrict__ outbf) {
  int i = blockIdx.x * blockDim.x + threadIdx.x;
  if (i >= Bsz * Hm) return;
  int b = i >> 10;
  float v = hsum[i] * rem[b] * (1.f / (float)Tm);
  outp[i] = v; outbf[i] = (__bf16)v;
}

// ---------------- gate logits + top2 + softmax + expert dispatch ----------------
__global__ __launch_bounds__(256) void k_gate_topk(
    const float* __restrict__ outp, const float* __restrict__ gw,
    const float* __restrict__ gb, int* __restrict__ idx,
    float* __restrict__ wts, int* __restrict__ counts,
    int* __restrict__ rowmap) {
  int b = blockIdx.x * 8 + (threadIdx.x >> 5);
  int lane = threadIdx.x & 31;
  if (b >= Bsz) return;
  float acc[Em];
#pragma unroll
  for (int e = 0; e < Em; ++e) acc[e] = 0.f;
  for (int d = lane; d < Hm; d += 32) {
    float x = outp[(size_t)b * Hm + d];
#pragma unroll
    for (int e = 0; e < Em; ++e) acc[e] += x * gw[d * Em + e];
  }
#pragma unroll
  for (int e = 0; e < Em; ++e)
#pragma unroll
    for (int off = 16; off; off >>= 1) acc[e] += __shfl_xor(acc[e], off, 32);
  if (lane == 0) {
    float lg[Em];
#pragma unroll
    for (int e = 0; e < Em; ++e) lg[e] = acc[e] + gb[e];
    int i0 = 0;
#pragma unroll
    for (int e = 1; e < Em; ++e) if (lg[e] > lg[i0]) i0 = e;
    int i1 = (i0 == 0) ? 1 : 0;
#pragma unroll
    for (int e = 0; e < Em; ++e) if (e != i0 && lg[e] > lg[i1]) i1 = e;
    float e1 = __expf(lg[i1] - lg[i0]);
    float inv = 1.f / (1.f + e1);
    idx[b * 2] = i0; idx[b * 2 + 1] = i1;
    wts[b * 2] = inv; wts[b * 2 + 1] = e1 * inv;
    int p0 = atomicAdd(&counts[i0], 1);
    rowmap[i0 * (Bsz * Km) + p0] = b * 2;
    int p1 = atomicAdd(&counts[i1], 1);
    rowmap[i1 * (Bsz * Km) + p1] = b * 2 + 1;
  }
}

// ---------------- weighted top-2 combine ----------------
__global__ __launch_bounds__(256) void k_combine(
    const float* __restrict__ sel, const float* __restrict__ wts,
    float* __restrict__ outp, __bf16* __restrict__ outbf) {
  int i = blockIdx.x * blockDim.x + threadIdx.x;
  if (i >= Bsz * Om) return;
  int b = i >> 10, o = i & (Om - 1);
  float v = wts[b * 2]     * sel[(size_t)(b * 2) * Om + o] +
            wts[b * 2 + 1] * sel[(size_t)(b * 2 + 1) * Om + o];
  outp[i] = v; outbf[i] = (__bf16)v;
}

// ---------------- LayerNorm ----------------
__global__ __launch_bounds__(256) void k_layernorm(
    const float* __restrict__ outp, const float* __restrict__ g,
    const float* __restrict__ bb, float* __restrict__ y) {
  int row = blockIdx.x;
  const float* pr = outp + (size_t)row * Om;
  float s = 0.f, s2 = 0.f;
  for (int o = threadIdx.x; o < Om; o += 256) {
    float v = pr[o]; s += v; s2 += v * v;
  }
  __shared__ float rs[8], rs2[8];
  int lane = threadIdx.x & 31, wv = threadIdx.x >> 5;
#pragma unroll
  for (int off = 16; off; off >>= 1) {
    s += __shfl_xor(s, off, 32); s2 += __shfl_xor(s2, off, 32);
  }
  if (lane == 0) { rs[wv] = s; rs2[wv] = s2; }
  __syncthreads();
  if (wv == 0) {
    s  = (lane < 8) ? rs[lane]  : 0.f;
    s2 = (lane < 8) ? rs2[lane] : 0.f;
#pragma unroll
    for (int off = 4; off; off >>= 1) {
      s += __shfl_xor(s, off, 32); s2 += __shfl_xor(s2, off, 32);
    }
    if (lane == 0) { rs[0] = s; rs2[0] = s2; }
  }
  __syncthreads();
  float mu  = rs[0] * (1.f / (float)Om);
  float var = rs2[0] * (1.f / (float)Om) - mu * mu;
  float inv = rsqrtf(var + 1e-5f);
  for (int o = threadIdx.x; o < Om; o += 256)
    y[(size_t)row * Om + o] = (pr[o] - mu) * inv * g[o] + bb[o];
}

// ---------------- host orchestration ----------------
extern "C" void kernel_launch(void* const* d_in, const int* in_sizes, int n_in,
                              void* d_out, int out_size, void* d_ws, size_t ws_size,
                              hipStream_t stream) {
  const float* x        = (const float*)d_in[0];
  const float* lstm_wih = (const float*)d_in[1];
  const float* lstm_whh = (const float*)d_in[2];
  const float* lstm_bih = (const float*)d_in[3];
  const float* lstm_bhh = (const float*)d_in[4];
  const float* halt_w   = (const float*)d_in[5];
  const float* halt_b   = (const float*)d_in[6];
  const float* gate_w   = (const float*)d_in[7];
  const float* gate_b   = (const float*)d_in[8];
  const float* w1       = (const float*)d_in[9];
  const float* b1       = (const float*)d_in[10];
  const float* w2       = (const float*)d_in[11];
  const float* b2       = (const float*)d_in[12];
  const float* ln_g     = (const float*)d_in[13];
  const float* ln_b     = (const float*)d_in[14];
  (void)in_sizes; (void)n_in; (void)out_size; (void)ws_size;

  size_t off = 0;
  auto carve = [&](size_t bytes) -> char* {
    char* p = (char*)d_ws + off;
    off += (bytes + 255) & ~(size_t)255;
    return p;
  };
  const size_t BH = (size_t)Bsz * Hm;
  __bf16* x_bf    = (__bf16*)carve((size_t)Bsz * Dm * 2);
  __bf16* wih_bf  = (__bf16*)carve((size_t)4 * Hm * Dm * 2);
  __bf16* whh_bf  = (__bf16*)carve((size_t)4 * Hm * Hm * 2);
  __bf16* w1_bf   = (__bf16*)carve((size_t)Lm * Em * Hm * Hm * 2);
  __bf16* w2_bf   = (__bf16*)carve((size_t)Lm * Em * Hm * Om * 2);
  float*  zX      = (float*)carve((size_t)Bsz * 4 * Hm * 4);
  float*  zH      = (float*)carve((size_t)Bsz * 4 * Hm * 4);
  float*  hbuf    = (float*)carve(BH * 4);
  float*  cbuf    = (float*)carve(BH * 4);
  float*  hsum    = (float*)carve(BH * 4);
  __bf16* h_bf    = (__bf16*)carve(BH * 2);
  float*  haltb   = (float*)carve((size_t)Bsz * 4);
  float*  remb    = (float*)carve((size_t)Bsz * 4);
  float*  out_f   = (float*)carve(BH * 4);
  __bf16* out_bf  = (__bf16*)carve(BH * 2);
  int*    idxb    = (int*)carve((size_t)Bsz * Km * 4);
  float*  wtsb    = (float*)carve((size_t)Bsz * Km * 4);
  int*    counts  = (int*)carve((size_t)Em * 4);
  int*    rowmap  = (int*)carve((size_t)Em * Bsz * Km * 4);
  __bf16* hid_sel = (__bf16*)carve((size_t)Bsz * Km * Hm * 2);
  float*  selb    = (float*)carve((size_t)Bsz * Km * Om * 4);

  // zero ACT state (h, c, h_sum, halt, rem, h_bf)
  hipMemsetAsync(hbuf, 0, BH * 4, stream);
  hipMemsetAsync(cbuf, 0, BH * 4, stream);
  hipMemsetAsync(hsum, 0, BH * 4, stream);
  hipMemsetAsync(h_bf, 0, BH * 2, stream);
  hipMemsetAsync(haltb, 0, (size_t)Bsz * 4, stream);
  hipMemsetAsync(remb, 0, (size_t)Bsz * 4, stream);

  // fp32 -> bf16 conversions
  auto cvt = [&](const float* s, __bf16* d, long long n) {
    k_cvt_bf16<<<2048, 256, 0, stream>>>(s, d, n / 4);
  };
  cvt(x,        x_bf,   (long long)Bsz * Dm);
  cvt(lstm_wih, wih_bf, (long long)4 * Hm * Dm);
  cvt(lstm_whh, whh_bf, (long long)4 * Hm * Hm);
  cvt(w1,       w1_bf,  (long long)Lm * Em * Hm * Hm);
  cvt(w2,       w2_bf,  (long long)Lm * Em * Hm * Om);

  // zX = x @ Wih^T  (computed ONCE: invariant over T)
  {
    GemmArgs ga = {};
    ga.A = x_bf; ga.lda = Dm;
    ga.Bm = wih_bf; ga.ldb = Dm; ga.bT = 1;
    ga.bias = nullptr; ga.C = zX; ga.ldc = 4 * Hm; ga.cBf16 = 0; ga.relu = 0;
    ga.M = Bsz; ga.N = 4 * Hm; ga.Kd = Dm;
    k_gemm_wmma<<<dim3((4 * Hm) / TBN, Bsz / TBM, 1), 256, 0, stream>>>(ga);
  }

  // T LSTM steps
  for (int t = 0; t < Tm; ++t) {
    GemmArgs ga = {};
    ga.A = h_bf; ga.lda = Hm;
    ga.Bm = whh_bf; ga.ldb = Hm; ga.bT = 1;
    ga.C = zH; ga.ldc = 4 * Hm; ga.cBf16 = 0; ga.relu = 0;
    ga.M = Bsz; ga.N = 4 * Hm; ga.Kd = Hm;
    k_gemm_wmma<<<dim3((4 * Hm) / TBN, Bsz / TBM, 1), 256, 0, stream>>>(ga);
    k_lstm_cell<<<(int)(BH / 256), 256, 0, stream>>>(zX, zH, lstm_bih, lstm_bhh,
                                                     cbuf, hbuf, h_bf, hsum);
    k_halt<<<Bsz / 8, 256, 0, stream>>>(hbuf, halt_w, halt_b, haltb, remb);
  }
  k_act_final<<<(int)(BH / 256), 256, 0, stream>>>(hsum, remb, out_f, out_bf);

  // MoE levels with sparse top-2 dispatch
  for (int l = 0; l < Lm; ++l) {
    hipMemsetAsync(counts, 0, Em * 4, stream);
    k_gate_topk<<<Bsz / 8, 256, 0, stream>>>(out_f, gate_w + (size_t)l * Hm * Em,
                                             gate_b + (size_t)l * Em,
                                             idxb, wtsb, counts, rowmap);
    // expert GEMM 1: hid = relu(out @ w1[l,e] + b1[l,e])  (gathered rows)
    {
      GemmArgs ga = {};
      ga.A = out_bf; ga.lda = Hm; ga.aShift = 1;           // row = g >> 1
      ga.Bm = w1_bf + (size_t)l * Em * Hm * Hm; ga.ldb = Hm; ga.bT = 0;
      ga.bExpStride = (long long)Hm * Hm;
      ga.bias = b1 + (size_t)l * Em * Hm; ga.biasExpStride = Hm;
      ga.map = rowmap; ga.mapStride = Bsz * Km; ga.counts = counts;
      ga.C = hid_sel; ga.ldc = Hm; ga.cBf16 = 1; ga.relu = 1;
      ga.M = Bsz * Km; ga.N = Hm; ga.Kd = Hm;
      k_gemm_wmma<<<dim3(Hm / TBN, (Bsz * Km) / TBM, Em), 256, 0, stream>>>(ga);
    }
    // expert GEMM 2: allo = hid @ w2[l,e] + b2[l,e]
    {
      GemmArgs ga = {};
      ga.A = hid_sel; ga.lda = Hm; ga.aShift = 0;          // row = g
      ga.Bm = w2_bf + (size_t)l * Em * Hm * Om; ga.ldb = Om; ga.bT = 0;
      ga.bExpStride = (long long)Hm * Om;
      ga.bias = b2 + (size_t)l * Em * Om; ga.biasExpStride = Om;
      ga.map = rowmap; ga.mapStride = Bsz * Km; ga.counts = counts;
      ga.C = selb; ga.ldc = Om; ga.cBf16 = 0; ga.relu = 0;
      ga.M = Bsz * Km; ga.N = Om; ga.Kd = Hm;
      k_gemm_wmma<<<dim3(Om / TBN, (Bsz * Km) / TBM, Em), 256, 0, stream>>>(ga);
    }
    k_combine<<<(int)((size_t)Bsz * Om / 256), 256, 0, stream>>>(selb, wtsb,
                                                                 out_f, out_bf);
  }

  k_layernorm<<<Bsz, 256, 0, stream>>>(out_f, ln_g, ln_b, (float*)d_out);
}
